// SupConLoss_63419487093292
// MI455X (gfx1250) — compile-verified, matrix operand order checked by reading, not run on hardware
//
#include <hip/hip_runtime.h>

typedef __attribute__((ext_vector_type(16))) _Float16 v16h;
typedef __attribute__((ext_vector_type(8)))  _Float16 v8h;
typedef __attribute__((ext_vector_type(8)))  float    v8f;

#define NN      4096
#define DD      256
#define KSEL    128
#define TEMP    0.1f
#define BASETMP 0.07f
#define GSTRIDE 4104   /* 4096 + 8 floats pad: breaks 64-bank alignment between rows */

__device__ __forceinline__ float wave_max(float v) {
    #pragma unroll
    for (int m = 16; m; m >>= 1) v = fmaxf(v, __shfl_xor(v, m, 32));
    return v;
}
__device__ __forceinline__ float wave_sum(float v) {
    #pragma unroll
    for (int m = 16; m; m >>= 1) v += __shfl_xor(v, m, 32);
    return v;
}
__device__ __forceinline__ int wave_sumi(int v) {
    #pragma unroll
    for (int m = 16; m; m >>= 1) v += __shfl_xor(v, m, 32);
    return v;
}
// Map f32 to monotone-ordered u32 key (total order matching float <)
__device__ __forceinline__ unsigned fkey(float x) {
    unsigned u = __float_as_uint(x);
    return (u >> 31) ? ~u : (u | 0x80000000u);
}

// Load a 16-row x 32-K f16 fragment in the CDNA5 WMMA A/B VGPR layout:
// lane L holds matrix row (row0 + L%16); half h<8 -> K = k0 + (L/16)*8 + h,
// half h>=8 -> K = k0 + 16 + (L/16)*8 + (h-8). Two 16-byte loads per lane.
__device__ __forceinline__ v16h load_frag(const _Float16* __restrict__ f,
                                          int row0, int k0, int lane) {
    int r    = row0 + (lane & 15);
    int base = k0 + ((lane >> 4) << 3);
    const v8h* p = (const v8h*)(f + r * DD + base);
    v8h a = p[0];   // K = base .. base+7
    v8h b = p[2];   // K = base+16 .. base+23
    return __builtin_shufflevector(a, b, 0,1,2,3,4,5,6,7,8,9,10,11,12,13,14,15);
}

// ---------------- Kernel 1: L2-normalize rows, emit f16 ----------------
__global__ void supcon_norm(const float* __restrict__ feat, _Float16* __restrict__ fn) {
    int wave = threadIdx.x >> 5, lane = threadIdx.x & 31;
    int row  = blockIdx.x * 8 + wave;
    const float* src = feat + row * DD;
    float v[8];
    float s = 0.f;
    #pragma unroll
    for (int i = 0; i < 8; ++i) { v[i] = src[lane + i * 32]; s += v[i] * v[i]; }
    s = wave_sum(s);
    float inv = 1.0f / sqrtf(s);
    #pragma unroll
    for (int i = 0; i < 8; ++i) fn[row * DD + lane + i * 32] = (_Float16)(v[i] * inv);
}

// ---------------- Kernel 2: Gram stripe (WMMA) + per-row SupCon math ----------------
__global__ void supcon_main(const _Float16* __restrict__ fn,
                            const long long* __restrict__ labels,
                            float* __restrict__ loss) {
    extern __shared__ unsigned char smem[];
    float*         G   = (float*)smem;                            // 16 x GSTRIDE
    unsigned char* lab = smem + (size_t)16 * GSTRIDE * sizeof(float); // NN bytes

    int tid = threadIdx.x, wave = tid >> 5, lane = tid & 31;
    int row0 = blockIdx.x * 16;

    // Cache expanded labels (concat of two halves) as bytes in LDS
    for (int j = tid; j < NN; j += 256) lab[j] = (unsigned char)labels[j & (NN / 2 - 1)];

    // Preload the 16 A-rows for this block (all 256 K) into registers
    v16h A[8];
    #pragma unroll
    for (int kc = 0; kc < 8; ++kc) A[kc] = load_frag(fn, row0, kc * 32, lane);

    int hi = lane >> 4, lo = lane & 15;
    // Each wave computes columns [wave*512, wave*512+512) as 32 WMMA tiles
    for (int t = 0; t < 32; ++t) {
        int n0 = wave * 512 + t * 16;
        v8f c = {};
        #pragma unroll
        for (int kc = 0; kc < 8; ++kc) {
            v16h B = load_frag(fn, n0, kc * 32, lane);
            c = __builtin_amdgcn_wmma_f32_16x16x32_f16(false, A[kc], false, B,
                                                       (short)0, c, false, false);
        }
        // C layout: lane L -> N = n0 + L%16, VGPR r -> M = r + 8*(L/16)
        #pragma unroll
        for (int r = 0; r < 8; ++r)
            G[(r + 8 * hi) * GSTRIDE + n0 + lo] = c[r];
    }
    __syncthreads();

    const float invT = 1.0f / TEMP;
    // 2 rows per wave
    for (int m = wave; m < 16; m += 8) {
        const float* row = G + m * GSTRIDE;
        int gi = row0 + m;
        int li = lab[gi];

        // 1) row max of G (logits_max = max/T, handled by subtracting in G-space)
        float mx = -3.4e38f;
        for (int j = lane; j < NN; j += 32) mx = fmaxf(mx, row[j]);
        mx = wave_max(mx);

        // 2) sum of exp(logits) excluding the diagonal
        float s = 0.f;
        for (int j = lane; j < NN; j += 32)
            if (j != gi) s += __expf((row[j] - mx) * invT);
        s = wave_sum(s);
        float logZ = __logf(s);

        // 3) 128th-smallest dot product (== 128 farthest distances) by key bisection
        unsigned klo = 0u, khi = 0xFFFFFFFFu;
        while (klo < khi) {
            unsigned mid = klo + ((khi - klo) >> 1);
            int c = 0;
            for (int j = lane; j < NN; j += 32) c += (fkey(row[j]) <= mid);
            c = wave_sumi(c);
            if (c >= KSEL) khi = mid; else klo = mid + 1;
        }
        unsigned x = klo;  // 128th smallest key
        int c1 = 0;
        for (int j = lane; j < NN; j += 32) c1 += (fkey(row[j]) < x);
        c1 = wave_sumi(c1);
        int need = KSEL - c1;

        // 4) accumulate log_prob over selected & same-label columns (ties by index order)
        float acc = 0.f;
        int cnt = 0, ties = 0;
        for (int cc = 0; cc < NN / 32; ++cc) {
            int j = cc * 32 + lane;
            float g = row[j];
            unsigned k = fkey(g);
            unsigned m32 = (unsigned)__ballot(k == x);
            int pre = __popc(m32 & ((1u << lane) - 1u));
            bool sel = (k < x) || ((k == x) && (ties + pre < need));
            ties += __popc(m32);
            if (sel && j != gi && (int)lab[j] == li) {
                acc += (g - mx) * invT - logZ;
                cnt++;
            }
        }
        acc = wave_sum(acc);
        cnt = wave_sumi(cnt);
        if (lane == 0)
            loss[gi] = -(TEMP / BASETMP) * (acc / (float)cnt);
    }
}

// ---------------- Kernel 3: mean over per-row losses ----------------
__global__ void supcon_reduce(const float* __restrict__ loss, float* __restrict__ out) {
    __shared__ float sm[8];
    int tid = threadIdx.x;
    float s = 0.f;
    for (int j = tid; j < NN; j += 256) s += loss[j];
    s = wave_sum(s);
    if ((tid & 31) == 0) sm[tid >> 5] = s;
    __syncthreads();
    if (tid < 8) {
        float v = sm[tid];
        #pragma unroll
        for (int m = 4; m; m >>= 1) v += __shfl_xor(v, m, 32);
        if (tid == 0) out[0] = v / (float)NN;
    }
}

extern "C" void kernel_launch(void* const* d_in, const int* in_sizes, int n_in,
                              void* d_out, int out_size, void* d_ws, size_t ws_size,
                              hipStream_t stream) {
    const float*     feat   = (const float*)d_in[0];
    const long long* labels = (const long long*)d_in[1];
    // n_views (d_in[2]) == 1: mean over all rows, ignored.

    _Float16* fn   = (_Float16*)d_ws;                                   // 2 MB
    float*    loss = (float*)((char*)d_ws + (size_t)NN * DD * sizeof(_Float16)); // 16 KB
    float*    out  = (float*)d_out;

    supcon_norm<<<NN / 8, 256, 0, stream>>>(feat, fn);

    size_t smem = (size_t)16 * GSTRIDE * sizeof(float) + NN;  // 266,752 B < 320 KB WGP LDS
    supcon_main<<<NN / 16, 256, smem, stream>>>(fn, labels, loss);

    supcon_reduce<<<1, 256, 0, stream>>>(loss, out);
}
